// SpikeSelfAttention_180388627113
// MI455X (gfx1250) — matrix operand order repaired and checked by reading, not compiled, not run on hardware
//
#include <hip/hip_runtime.h>
#include <hip/hip_bf16.h>

// ---------------------------------------------------------------------------
// Types for CDNA5 WMMA / TDM
// ---------------------------------------------------------------------------
typedef __attribute__((ext_vector_type(16))) __bf16    v16bf;
typedef __attribute__((ext_vector_type(8)))  __bf16    v8bf;
typedef __attribute__((ext_vector_type(16))) _Float16  v16h;
typedef __attribute__((ext_vector_type(8)))  _Float16  v8h;
typedef __attribute__((ext_vector_type(8)))  float     v8f;
typedef __attribute__((ext_vector_type(4)))  unsigned  v4u;
typedef __attribute__((ext_vector_type(8)))  int       v8i;
typedef __attribute__((ext_vector_type(4)))  int       v4i;

#define DIM       768
#define NHEAD     12
#define HDIM      64
#define BATCH     8
#define SEQ       1024
#define ROWS      (BATCH * SEQ)        // 8192
#define OUT3C     (3 * DIM)            // 2304

#if defined(__has_builtin)
# if __has_builtin(__builtin_amdgcn_tensor_load_to_lds) && __has_builtin(__builtin_amdgcn_s_wait_tensorcnt)
#  define USE_TDM 1
# endif
#endif
#ifndef USE_TDM
# define USE_TDM 0
#endif
#if defined(__has_include)
# if __has_include(<hip/amd_detail/amd_gfx1250_TDM.h>)
#  define TDM_SIX_ARGS 1           // therock-10.0 headers -> 6-arg builtin
# endif
#endif

// ---------------------------------------------------------------------------
// WMMA fragment loaders (wave32; A/B use contiguous k, same in both, not the ISA table order)
// A (16x32, 16-bit): lane l<16 -> M=l, K {k0+0..15};
//                    lane l>=16 -> M=l-16, K {k0+16..31}.
// B (32x16, 16-bit): lane l -> N=l%16, 16 contiguous K from k0+(l<16?0:16).
// C/D (16x16 f32):   slot r, lane l -> M = r + (l<16?0:8), N = l%16.
// ---------------------------------------------------------------------------
__device__ inline v16bf load_a_bf(const __bf16* __restrict__ row, int k0, int hl) {
  v8bf lo = *(const v8bf*)(row + k0 + hl * 16);
  v8bf hi = *(const v8bf*)(row + k0 + hl * 16 + 8);
  v16bf a;
#pragma unroll
  for (int i = 0; i < 8; ++i) { a[i] = lo[i]; a[8 + i] = hi[i]; }
  return a;
}

__device__ inline v16h load_a_h(const _Float16* __restrict__ row, int k0, int hl) {
  v8h lo = *(const v8h*)(row + k0 + hl * 16);
  v8h hi = *(const v8h*)(row + k0 + hl * 16 + 8);
  v16h a;
#pragma unroll
  for (int i = 0; i < 8; ++i) { a[i] = lo[i]; a[8 + i] = hi[i]; }
  return a;
}

// ---------------------------------------------------------------------------
// Kernel 0: f32 -> bf16 conversion
// ---------------------------------------------------------------------------
__global__ __launch_bounds__(256) void cvt_bf16(const float* __restrict__ src,
                                                __bf16* __restrict__ dst, int n) {
  int i = blockIdx.x * 256 + threadIdx.x;
  if (i < n) dst[i] = (__bf16)src[i];
}

// ---------------------------------------------------------------------------
// Kernels 1/5: GEMM out[M,O] = X[M,C] * W[O,C]^T (+bias).
// One wave computes a 32x64 block: 8 WMMAs per K-step on 6 fragment loads.
// ---------------------------------------------------------------------------
template <int OCOLS, bool HAS_BIAS>
__global__ __launch_bounds__(256) void gemm_bf16(const __bf16* __restrict__ X,
                                                 const __bf16* __restrict__ W,
                                                 const float* __restrict__ bias,
                                                 float* __restrict__ out) {
  const int lane = threadIdx.x & 31;
  const int wave = threadIdx.x >> 5;
  const int col  = lane & 15;
  const int hl   = lane >> 4;
  const int OB   = OCOLS / 64;
  const int wt   = blockIdx.x * 8 + wave;
  const int tm   = wt / OB;          // 32-row block
  const int tn   = wt % OB;          // 64-col block

  const __bf16* a0p = X + (size_t)(tm * 32 + col) * DIM;
  const __bf16* a1p = a0p + (size_t)16 * DIM;
  const __bf16* bp[4];
#pragma unroll
  for (int t = 0; t < 4; ++t) bp[t] = W + (size_t)(tn * 64 + t * 16 + col) * DIM;

  v8f acc[2][4] = {};
  for (int k0 = 0; k0 < DIM; k0 += 32) {
    v16bf a0 = load_a_bf(a0p, k0, hl);
    v16bf a1 = load_a_bf(a1p, k0, hl);
#pragma unroll
    for (int t = 0; t < 4; ++t) {
      v16bf b = *(const v16bf*)(bp[t] + k0 + hl * 16);
      acc[0][t] = __builtin_amdgcn_wmma_f32_16x16x32_bf16(false, a0, false, b,
                                                          (short)0, acc[0][t], false, false);
      acc[1][t] = __builtin_amdgcn_wmma_f32_16x16x32_bf16(false, a1, false, b,
                                                          (short)0, acc[1][t], false, false);
    }
  }
#pragma unroll
  for (int s = 0; s < 2; ++s) {
#pragma unroll
    for (int t = 0; t < 4; ++t) {
      float bv = HAS_BIAS ? bias[tn * 64 + t * 16 + col] : 0.0f;
      float* op = out + (size_t)(tm * 32 + s * 16 + hl * 8) * OCOLS + tn * 64 + t * 16 + col;
#pragma unroll
      for (int r = 0; r < 8; ++r) op[(size_t)r * OCOLS] = acc[s][t][r] + bv;
    }
  }
}

// ---------------------------------------------------------------------------
// Kernel 2: per-channel mean / rstd (train-mode BN, biased variance)
// ---------------------------------------------------------------------------
__global__ __launch_bounds__(256) void bn_stats(const float* __restrict__ qkv,
                                                float* __restrict__ meanOut,
                                                float* __restrict__ rstdOut) {
  const int c = blockIdx.x;
  float s = 0.f, s2 = 0.f;
  for (int r = threadIdx.x; r < ROWS; r += 256) {
    float v = qkv[(size_t)r * OUT3C + c];
    s += v; s2 += v * v;
  }
  __shared__ float sh[256], sh2[256];
  sh[threadIdx.x] = s; sh2[threadIdx.x] = s2;
  __syncthreads();
  for (int st = 128; st > 0; st >>= 1) {
    if (threadIdx.x < st) {
      sh[threadIdx.x]  += sh[threadIdx.x + st];
      sh2[threadIdx.x] += sh2[threadIdx.x + st];
    }
    __syncthreads();
  }
  if (threadIdx.x == 0) {
    float m   = sh[0]  * (1.0f / ROWS);
    float var = sh2[0] * (1.0f / ROWS) - m * m;
    meanOut[c] = m;
    rstdOut[c] = rsqrtf(var + 1e-5f);
  }
}

// ---------------------------------------------------------------------------
// Kernel 3: fused BN + MultiStepLIF (tau=2, v_th=1, hard reset) + spike pack.
// q spikes pre-scaled by SCALE = 1/8 (exact in f16)  -> [B,H,N,D]
// k spikes                                           -> [B,H,N,D]
// v spikes stored TRANSPOSED                          -> [B,H,D,N]
// ---------------------------------------------------------------------------
__global__ __launch_bounds__(256) void bn_lif_pack(const float* __restrict__ qkv,
                                                   const float* __restrict__ mean,
                                                   const float* __restrict__ rstd,
                                                   const float* __restrict__ gamma,
                                                   const float* __restrict__ beta,
                                                   _Float16* __restrict__ qs,
                                                   _Float16* __restrict__ ks,
                                                   _Float16* __restrict__ vt) {
  const int idx = blockIdx.x * 256 + threadIdx.x;   // over SEQ * OUT3C
  if (idx >= SEQ * OUT3C) return;
  const int c = idx % OUT3C;
  const int n = idx / OUT3C;
  const float m  = mean[c];
  const float rs = rstd[c];
  const float g  = gamma[c];
  const float bt = beta[c];

  const int which = c / DIM;
  const int cc = c % DIM;
  const int h = cc / HDIM;
  const int d = cc % HDIM;

  float vmem = 0.f;
#pragma unroll
  for (int b = 0; b < BATCH; ++b) {
    float x = qkv[(size_t)(b * SEQ + n) * OUT3C + c];
    x = (x - m) * rs * g + bt;               // batchnorm
    vmem += (x - vmem) * 0.5f;               // charge, tau = 2
    float s = (vmem >= 1.0f) ? 1.0f : 0.0f;  // fire
    vmem *= (1.0f - s);                      // hard reset
    const int bh = b * NHEAD + h;
    if (which == 0)
      qs[((size_t)bh * SEQ + n) * HDIM + d] = (_Float16)(s * 0.125f); // * SCALE
    else if (which == 1)
      ks[((size_t)bh * SEQ + n) * HDIM + d] = (_Float16)s;
    else
      vt[((size_t)bh * HDIM + d) * SEQ + n] = (_Float16)s;            // transposed
  }
}

// ---------------------------------------------------------------------------
// TDM: stage a 32x64-half K tile (4 KB, row-major) into LDS.
// 2D descriptor: data_size=2B, tile 64x32, tensor_dim0_stride=64.
// ---------------------------------------------------------------------------
#if USE_TDM
__device__ inline void tdm_load_k(const _Float16* gsrc, unsigned ldsByteAddr) {
  unsigned long long ga = (unsigned long long)(size_t)gsrc;
  v4u g0;
  g0[0] = 1u;                                   // count=1, user mode
  g0[1] = ldsByteAddr;                          // lds_addr
  g0[2] = (unsigned)ga;                         // global_addr[31:0]
  g0[3] = (unsigned)((ga >> 32) & 0x1FFFFFFu) | 0x80000000u;  // addr[56:32] | type=2
  v8i g1;
  g1[0] = 0x00010000;        // workgroup_mask=0, data_size=1 (2B)
  g1[1] = (64 << 16);        // tensor_dim0 = 64  (bits 79:48)
  g1[2] = (32 << 16);        // tensor_dim1 = 32  (bits 111:80)
  g1[3] = (64 << 16);        // tile_dim0 = 64    (bits 127:112)
  g1[4] = 32;                // tile_dim1 = 32, tile_dim2 = 0
  g1[5] = 64;                // tensor_dim0_stride = 64 (bits 207:160)
  g1[6] = 0;
  g1[7] = 0;
  v4i z = {0, 0, 0, 0};
#if defined(TDM_SIX_ARGS)
  v8i z8 = {0, 0, 0, 0, 0, 0, 0, 0};
  __builtin_amdgcn_tensor_load_to_lds(g0, g1, z, z, z8, 0);
#else
  __builtin_amdgcn_tensor_load_to_lds(g0, g1, z, z, 0);
#endif
}
#endif

// ---------------------------------------------------------------------------
// Kernel 4: flash attention, fixed-max softmax (spike scores are in [0, 8]).
// grid = (B*H, SEQ/128); 8 waves/block, each wave owns 16 query rows.
//   S   = Q Kt      : 4 x wmma_f16 (K staged in LDS by the TDM, double-buffered)
//   P   = exp(S-8)  : exact softmax shift (no running max / rescale needed)
//   O^T = V^T P^T   : 4 x wmma_f16 (V^T read contiguously from global;
//                     P^T bounced through per-wave LDS for the layout change)
//   denom: lane-local accumulation + ONE shfl_xor(16) at the end.
// ---------------------------------------------------------------------------
__global__ __launch_bounds__(256) void flash_attn(const _Float16* __restrict__ Q,
                                                  const _Float16* __restrict__ K,
                                                  const _Float16* __restrict__ Vt,
                                                  __bf16* __restrict__ AO) {
  __shared__ __align__(16) _Float16 kbuf[2][32 * 64];   // K tile double buffer
  __shared__ __align__(16) _Float16 pbuf[8][32 * 16];   // per-wave P^T bounce

  const int lane = threadIdx.x & 31;
  const int wave = threadIdx.x >> 5;
  const int col  = lane & 15;
  const int hl   = lane >> 4;

  const int bh = blockIdx.x;                 // 0..95
  const int b  = bh / NHEAD;
  const int h  = bh % NHEAD;
  const int row0 = blockIdx.y * 128 + wave * 16;

  const _Float16* Qb  = Q  + (size_t)bh * SEQ * HDIM;
  const _Float16* Kb  = K  + (size_t)bh * SEQ * HDIM;
  const _Float16* Vtb = Vt + (size_t)bh * HDIM * SEQ;

  // Q tile A-fragments (rows row0..row0+15, D split into two K=32 chunks)
  const _Float16* qrow = Qb + (size_t)(row0 + col) * HDIM;
  const v16h qa0 = load_a_h(qrow, 0, hl);
  const v16h qa1 = load_a_h(qrow, 32, hl);

  v8f ot[4] = {};          // O^T accumulator: 4 d-groups x (16 d x 16 m)
  float lacc = 0.f;        // softmax denominator, lane-local half-sum
  _Float16* sl = pbuf[wave];

#if USE_TDM
  if (wave == 0) tdm_load_k(Kb, (unsigned)(size_t)&kbuf[0][0]);
#else
  {
    const v8h* src = (const v8h*)Kb;
    ((v8h*)kbuf[0])[threadIdx.x] = src[threadIdx.x];   // 256 x 16B = 4 KB
  }
#endif

  int parity = 0;
  for (int n0 = 0; n0 < SEQ; n0 += 32) {
#if USE_TDM
    if (wave == 0) {
      if (n0 + 32 < SEQ) {
        tdm_load_k(Kb + (size_t)(n0 + 32) * HDIM, (unsigned)(size_t)&kbuf[parity ^ 1][0]);
        __builtin_amdgcn_s_wait_tensorcnt((short)1);   // K(n0) complete
      } else {
        __builtin_amdgcn_s_wait_tensorcnt((short)0);
      }
    }
#endif
    __syncthreads();                                   // K(n0) visible to all

    // ---- S = Q K^T over LDS-staged K tile ---------------------------------
    const _Float16* kb = kbuf[parity];
    v16h b00 = *(const v16h*)(kb + col * 64 + hl * 16);
    v16h b01 = *(const v16h*)(kb + col * 64 + 32 + hl * 16);
    v16h b10 = *(const v16h*)(kb + (col + 16) * 64 + hl * 16);
    v16h b11 = *(const v16h*)(kb + (col + 16) * 64 + 32 + hl * 16);
    v8f s0 = {}, s1 = {};
    s0 = __builtin_amdgcn_wmma_f32_16x16x32_f16(false, qa0, false, b00, (short)0, s0, false, false);
    s0 = __builtin_amdgcn_wmma_f32_16x16x32_f16(false, qa1, false, b01, (short)0, s0, false, false);
    s1 = __builtin_amdgcn_wmma_f32_16x16x32_f16(false, qa0, false, b10, (short)0, s1, false, false);
    s1 = __builtin_amdgcn_wmma_f32_16x16x32_f16(false, qa1, false, b11, (short)0, s1, false, false);

    // ---- P = exp(S - 8): store transposed (P^T[k][m]) into the wave slab --
#pragma unroll
    for (int r = 0; r < 8; ++r) {
      const int m = r + hl * 8;
      sl[col * 16 + m]        = (_Float16)__expf(s0[r] - 8.0f);
      sl[(col + 16) * 16 + m] = (_Float16)__expf(s1[r] - 8.0f);
    }
    __syncthreads();

    // ---- B-fragment of P^T (lane: col m, 16 contiguous k) + denominator ---
    v16h pb;
    float psum = 0.f;
#pragma unroll
    for (int i = 0; i < 16; ++i) {
      _Float16 pv = sl[(hl * 16 + i) * 16 + col];
      pb[i] = pv;
      psum += (float)pv;
    }
    lacc += psum;

    // ---- O^T += V^T P^T : A-fragments contiguous from global V^T ----------
#pragma unroll
    for (int g = 0; g < 4; ++g) {
      const _Float16* vrow = Vtb + (size_t)(g * 16 + col) * SEQ;
      v16h va = load_a_h(vrow, n0, hl);
      ot[g] = __builtin_amdgcn_wmma_f32_16x16x32_f16(false, va, false, pb,
                                                     (short)0, ot[g], false, false);
    }

#if !USE_TDM
    if (n0 + 32 < SEQ) {   // stage next K tile (prev readers are past barrier)
      const v8h* src = (const v8h*)(Kb + (size_t)(n0 + 32) * HDIM);
      ((v8h*)kbuf[parity ^ 1])[threadIdx.x] = src[threadIdx.x];
    }
#endif
    parity ^= 1;
  }

  // ---- finalize: single cross-half reduction for the denominator ----------
  const float l = lacc + __shfl_xor(lacc, 16, 32);
  const float rden = 1.0f / l;                  // l >= 1024*e^-8 > 0
  const size_t orow = (size_t)(b * SEQ + row0 + col) * DIM + h * HDIM;
#pragma unroll
  for (int g = 0; g < 4; ++g) {
    v8bf ov;
#pragma unroll
    for (int r = 0; r < 8; ++r) ov[r] = (__bf16)(ot[g][r] * rden);
    *(v8bf*)(AO + orow + g * 16 + hl * 8) = ov;   // 8 contiguous d per lane
  }
}

// ---------------------------------------------------------------------------
// Host launcher
// ---------------------------------------------------------------------------
static inline size_t ws_align(size_t x) { return (x + 255) & ~(size_t)255; }

extern "C" void kernel_launch(void* const* d_in, const int* in_sizes, int n_in,
                              void* d_out, int out_size, void* d_ws, size_t ws_size,
                              hipStream_t stream) {
  const float* x        = (const float*)d_in[0];   // [8, 1024, 768]
  const float* w_qkv    = (const float*)d_in[1];   // [2304, 768]
  const float* bn_gamma = (const float*)d_in[2];   // [2304]
  const float* bn_beta  = (const float*)d_in[3];   // [2304]
  const float* w_proj   = (const float*)d_in[4];   // [768, 768]
  const float* b_proj   = (const float*)d_in[5];   // [768]
  float* out = (float*)d_out;                      // [8, 1024, 768]

  char* ws = (char*)d_ws;
  size_t off = 0;
  auto carve = [&](size_t bytes) { void* p = ws + off; off = ws_align(off + bytes); return p; };

  __bf16*   xb    = (__bf16*)  carve((size_t)ROWS * DIM * 2);
  __bf16*   wqb   = (__bf16*)  carve((size_t)OUT3C * DIM * 2);
  __bf16*   wpb   = (__bf16*)  carve((size_t)DIM * DIM * 2);
  float*    qkv   = (float*)   carve((size_t)ROWS * OUT3C * 4);
  float*    meanv = (float*)   carve((size_t)OUT3C * 4);
  float*    rstdv = (float*)   carve((size_t)OUT3C * 4);
  _Float16* qs    = (_Float16*)carve((size_t)ROWS * DIM * 2);
  _Float16* ks    = (_Float16*)carve((size_t)ROWS * DIM * 2);
  _Float16* vt    = (_Float16*)carve((size_t)ROWS * DIM * 2);
  __bf16*   ao    = (__bf16*)  carve((size_t)ROWS * DIM * 2);
  (void)ws_size; (void)n_in; (void)in_sizes; (void)out_size;

  // 0) f32 -> bf16
  {
    int n = ROWS * DIM;
    cvt_bf16<<<(n + 255) / 256, 256, 0, stream>>>(x, xb, n);
    n = OUT3C * DIM;
    cvt_bf16<<<(n + 255) / 256, 256, 0, stream>>>(w_qkv, wqb, n);
    n = DIM * DIM;
    cvt_bf16<<<(n + 255) / 256, 256, 0, stream>>>(w_proj, wpb, n);
  }

  // 1) QKV GEMM: [8192,768] x [2304,768]^T -> [8192,2304] f32
  {
    const int waves = (ROWS / 32) * (OUT3C / 64);    // 256 * 36 = 9216
    gemm_bf16<OUT3C, false><<<waves / 8, 256, 0, stream>>>(xb, wqb, nullptr, qkv);
  }

  // 2) BN statistics per channel
  bn_stats<<<OUT3C, 256, 0, stream>>>(qkv, meanv, rstdv);

  // 3) BN + LIF + spike pack (V stored transposed)
  {
    const int n = SEQ * OUT3C;
    bn_lif_pack<<<(n + 255) / 256, 256, 0, stream>>>(qkv, meanv, rstdv,
                                                     bn_gamma, bn_beta, qs, ks, vt);
  }

  // 4) flash attention -> ao (bf16, [B, N, C])
  {
    dim3 grid(BATCH * NHEAD, SEQ / 128);
    flash_attn<<<grid, 256, 0, stream>>>(qs, ks, vt, ao);
  }

  // 5) projection GEMM + bias -> d_out (f32)
  {
    const int waves = (ROWS / 32) * (DIM / 64);      // 256 * 12 = 3072
    gemm_bf16<DIM, true><<<waves / 8, 256, 0, stream>>>(ao, wpb, b_proj, out);
  }
}
